// MultiHeadAttention_68229850464507
// MI455X (gfx1250) — compile-verified, matrix operand order checked by reading, not run on hardware
//
#include <hip/hip_runtime.h>
#include <hip/hip_bf16.h>

// ---------------------------------------------------------------------------
// MHA forward for gfx1250 (MI455X).
//   - All GEMMs: v_wmma_f32_16x16x32_f16 (f32 accumulate).
//   - Inputs/weights packed to f16 once; hot loops are pure b128 loads + WMMA.
//   - 2x2 register blocking (32x32 per wave) in the dense GEMMs.
//   - attn (1 GB, fp32) is materialized in d_out as the reference requires;
//     that HBM traffic (~4 GB total) is the roofline floor, compute is cheap.
// ---------------------------------------------------------------------------

#define BATCH   4
#define SEQ     2048
#define DMODEL  1024
#define NHEADS  16
#define DK      64
#define MROWS   (BATCH * SEQ)              // 8192
#define OUT0    ((size_t)MROWS * DMODEL)   // floats in region 0 of d_out

typedef __attribute__((ext_vector_type(16))) _Float16 v16h;
typedef __attribute__((ext_vector_type(8)))  float    v8f;

// A/B fragment (16x32 f16, ISA 7.12.2): lane holds row/col (lane&15), K half
// selected by (lane>>4)*8.  K offsets {0..7} and {16..23} are contiguous
// 16-byte runs -> two b128 loads.
union Frag128 { v16h v; uint4 q[2]; };

__device__ __forceinline__ v16h frag_f16(const _Float16* __restrict__ p) {
  Frag128 u;
  u.q[0] = *(const uint4*)(p + 0);
  u.q[1] = *(const uint4*)(p + 16);
  return u.v;
}

// f32 source fragment with in-flight conversion (used only for attn @ V).
__device__ __forceinline__ v16h frag_f32(const float* __restrict__ p) {
  v16h f;
#pragma unroll
  for (int g = 0; g < 4; ++g) {
    const int off = ((g & 1) ? 4 : 0) + ((g & 2) ? 16 : 0);   // 0,4,16,20
    float4 x = *(const float4*)(p + off);
    f[4 * g + 0] = (_Float16)x.x;
    f[4 * g + 1] = (_Float16)x.y;
    f[4 * g + 2] = (_Float16)x.z;
    f[4 * g + 3] = (_Float16)x.w;
  }
  return f;
}

__device__ __forceinline__ v8f wmma16(v16h a, v16h b, v8f c) {
  return __builtin_amdgcn_wmma_f32_16x16x32_f16(false, a, false, b, (short)0, c,
                                                false, false);
}

// ---------------------------------------------------------------------------
// Kernel 0: f32 -> f16 pack (vectorized: float4 in, 4xf16 out).
// ---------------------------------------------------------------------------
__global__ void mha_cvt_kernel(const float* __restrict__ src,
                               _Float16* __restrict__ dst, int n4) {
  const int i = blockIdx.x * blockDim.x + threadIdx.x;
  if (i < n4) {
    float4 x = ((const float4*)src)[i];
    union { _Float16 h[4]; uint2 u; } o;
    o.h[0] = (_Float16)x.x; o.h[1] = (_Float16)x.y;
    o.h[2] = (_Float16)x.z; o.h[3] = (_Float16)x.w;
    ((uint2*)dst)[i] = o.u;
  }
}

// Store one 16x16 f32 accumulator tile as head-split f16.
__device__ __forceinline__ void store_proj_tile(_Float16* __restrict__ out,
                                                v8f acc, int m0, int n0,
                                                int lane, int mode,
                                                const float* __restrict__ bias) {
  const int mr = lane & 15, half = lane >> 4;
  const int n = n0 + mr;
  const float bv = bias[n];
  const int h = n >> 6, dk = n & 63;
#pragma unroll
  for (int r = 0; r < 8; ++r) {
    const int m = m0 + r + half * 8;
    const int b = m >> 11, s = m & (SEQ - 1);
    const size_t idx = (mode == 0)
        ? ((size_t)((b * NHEADS + h) * SEQ + s)) * DK + dk
        : ((size_t)((b * NHEADS + h) * DK + dk)) * SEQ + s;
    out[idx] = (_Float16)(acc[r] + bv);
  }
}

// ---------------------------------------------------------------------------
// Kernel 1: projection GEMM  out = X @ W^T + bias (all-f16 operands).
// Wave computes 32x32 (2x2 register blocking).  Block = 8 waves (4M x 2N)
// -> block tile 128x64.  Grid = (DMODEL/64, MROWS/128).
// ---------------------------------------------------------------------------
__global__ void mha_proj_kernel(const _Float16* __restrict__ Xh,
                                const _Float16* __restrict__ Wh,
                                const float* __restrict__ bias,
                                _Float16* __restrict__ out, int mode) {
  const int lane = threadIdx.x & 31;
  const int wave = threadIdx.x >> 5;
  const int m0 = blockIdx.y * 128 + (wave >> 1) * 32;
  const int n0 = blockIdx.x * 64 + (wave & 1) * 32;
  const int mr = lane & 15;
  const int half = lane >> 4;

  const _Float16* a0 = Xh + (size_t)(m0 + mr) * DMODEL + half * 8;
  const _Float16* a1 = a0 + (size_t)16 * DMODEL;
  const _Float16* b0 = Wh + (size_t)(n0 + mr) * DMODEL + half * 8;  // B col n = W row n
  const _Float16* b1 = b0 + (size_t)16 * DMODEL;

  v8f c00 = {}, c01 = {}, c10 = {}, c11 = {};
#pragma unroll 2
  for (int k = 0; k < DMODEL; k += 32) {
    const v16h A0 = frag_f16(a0 + k), A1 = frag_f16(a1 + k);
    const v16h B0 = frag_f16(b0 + k), B1 = frag_f16(b1 + k);
    c00 = wmma16(A0, B0, c00);
    c01 = wmma16(A0, B1, c01);
    c10 = wmma16(A1, B0, c10);
    c11 = wmma16(A1, B1, c11);
  }
  store_proj_tile(out, c00, m0,      n0,      lane, mode, bias);
  store_proj_tile(out, c01, m0,      n0 + 16, lane, mode, bias);
  store_proj_tile(out, c10, m0 + 16, n0,      lane, mode, bias);
  store_proj_tile(out, c11, m0 + 16, n0 + 16, lane, mode, bias);
}

// Store one 16x16 scores tile (scaled + causal-masked).
__device__ __forceinline__ void store_score_tile(float* __restrict__ out,
                                                 v8f acc, int q0, int k0,
                                                 int lane) {
  const int mr = lane & 15, half = lane >> 4;
  const int n = k0 + mr;
  const float NEG_INF = -__builtin_inff();
#pragma unroll
  for (int r = 0; r < 8; ++r) {
    const int m = q0 + r + half * 8;
    float sc = acc[r] * 0.125f;          // 1/sqrt(64)
    if (n > m) sc = NEG_INF;
    out[(size_t)m * SEQ + n] = sc;
  }
}

__device__ __forceinline__ void store_inf_tile(float* __restrict__ out,
                                               int q0, int k0, int lane) {
  const int mr = lane & 15, half = lane >> 4;
  const int n = k0 + mr;
  const float NEG_INF = -__builtin_inff();
#pragma unroll
  for (int r = 0; r < 8; ++r)
    out[(size_t)(q0 + r + half * 8) * SEQ + n] = NEG_INF;
}

// ---------------------------------------------------------------------------
// Kernel 2: causal raw scores -> attn region (f32).  Wave tile 32x32 (2x2
// blocking, K = DK = 64).  Block 8 waves (4M x 2N) -> 128x64.
// Grid = (SEQ/64, SEQ/128, B*H).
// ---------------------------------------------------------------------------
__global__ void mha_scores_kernel(const _Float16* __restrict__ qh,
                                  const _Float16* __restrict__ kh,
                                  float* __restrict__ attn) {
  const int bh = blockIdx.z;
  const int lane = threadIdx.x & 31;
  const int wave = threadIdx.x >> 5;
  const int q0 = blockIdx.y * 128 + (wave >> 1) * 32;
  const int k0 = blockIdx.x * 64 + (wave & 1) * 32;
  float* out = attn + (size_t)bh * SEQ * SEQ;

  if (k0 > q0 + 31) {                    // 32x32 tile entirely above diagonal
    store_inf_tile(out, q0,      k0,      lane);
    store_inf_tile(out, q0,      k0 + 16, lane);
    store_inf_tile(out, q0 + 16, k0,      lane);
    store_inf_tile(out, q0 + 16, k0 + 16, lane);
    return;
  }

  const int mr = lane & 15;
  const int half = lane >> 4;
  const _Float16* a0 = qh + ((size_t)bh * SEQ + q0 + mr) * DK + half * 8;
  const _Float16* a1 = a0 + (size_t)16 * DK;
  const _Float16* b0 = kh + ((size_t)bh * SEQ + k0 + mr) * DK + half * 8;
  const _Float16* b1 = b0 + (size_t)16 * DK;

  v8f c00 = {}, c01 = {}, c10 = {}, c11 = {};
#pragma unroll
  for (int k = 0; k < DK; k += 32) {
    const v16h A0 = frag_f16(a0 + k), A1 = frag_f16(a1 + k);
    const v16h B0 = frag_f16(b0 + k), B1 = frag_f16(b1 + k);
    c00 = wmma16(A0, B0, c00);
    c01 = wmma16(A0, B1, c01);
    c10 = wmma16(A1, B0, c10);
    c11 = wmma16(A1, B1, c11);
  }
  store_score_tile(out, c00, q0,      k0,      lane);
  store_score_tile(out, c01, q0,      k0 + 16, lane);
  store_score_tile(out, c10, q0 + 16, k0,      lane);
  store_score_tile(out, c11, q0 + 16, k0 + 16, lane);
}

// ---------------------------------------------------------------------------
// Kernel 3: in-place row softmax.  One 256-thread block per row (8 elem/thr).
// ---------------------------------------------------------------------------
__global__ void mha_softmax_kernel(float* __restrict__ attn) {
  __shared__ float red[256];
  float* p = attn + (size_t)blockIdx.x * SEQ;
  const int t = threadIdx.x;

  float x[8];
  float mx = -__builtin_inff();
#pragma unroll
  for (int i = 0; i < 8; ++i) {
    x[i] = p[t + i * 256];
    mx = fmaxf(mx, x[i]);
  }
  red[t] = mx;
  __syncthreads();
  for (int s = 128; s > 0; s >>= 1) {
    if (t < s) red[t] = fmaxf(red[t], red[t + s]);
    __syncthreads();
  }
  mx = red[0];
  __syncthreads();

  float sum = 0.f;
#pragma unroll
  for (int i = 0; i < 8; ++i) {
    x[i] = __expf(x[i] - mx);            // exp(-inf) == 0 handles the mask
    sum += x[i];
  }
  red[t] = sum;
  __syncthreads();
  for (int s = 128; s > 0; s >>= 1) {
    if (t < s) red[t] += red[t + s];
    __syncthreads();
  }
  const float inv = 1.0f / red[0];
#pragma unroll
  for (int i = 0; i < 8; ++i) p[t + i * 256] = x[i] * inv;
}

// ---------------------------------------------------------------------------
// Kernel 4: ctx = attn @ v.  Wave tile 16M x 64N: one attn f32 A-fragment is
// reused across all 4 d_k column tiles (attn reads dominate HBM traffic).
// K-loop truncated at the causal boundary (masked probs are exactly 0).
// Block = 8 waves stacked in M -> 128 rows.  Grid = (1, SEQ/128, B*H).
// ---------------------------------------------------------------------------
__global__ void mha_ctx_kernel(const float* __restrict__ attn,
                               const _Float16* __restrict__ vT,
                               _Float16* __restrict__ ctxh) {
  const int bh = blockIdx.z;
  const int lane = threadIdx.x & 31;
  const int wave = threadIdx.x >> 5;
  const int q0 = blockIdx.y * 128 + wave * 16;
  const int mr = lane & 15;
  const int half = lane >> 4;

  const float* arow = attn + ((size_t)bh * SEQ + q0 + mr) * SEQ + half * 8;
  const _Float16* brow = vT + ((size_t)bh * DK + mr) * SEQ + half * 8;

  const int kend = ((q0 + 16) + 31) & ~31;   // rows q0..q0+15 valid for k<=q
  v8f c0 = {}, c1 = {}, c2 = {}, c3 = {};
  for (int k = 0; k < kend; k += 32) {
    const v16h A = frag_f32(arow + k);
    c0 = wmma16(A, frag_f16(brow + (size_t)0  * SEQ + k), c0);
    c1 = wmma16(A, frag_f16(brow + (size_t)16 * SEQ + k), c1);
    c2 = wmma16(A, frag_f16(brow + (size_t)32 * SEQ + k), c2);
    c3 = wmma16(A, frag_f16(brow + (size_t)48 * SEQ + k), c3);
  }

  const int b = bh >> 4, h = bh & 15;
  v8f cc[4] = {c0, c1, c2, c3};
#pragma unroll
  for (int nt = 0; nt < 4; ++nt) {
    const int n = nt * 16 + mr;
#pragma unroll
    for (int r = 0; r < 8; ++r) {
      const int q = q0 + r + half * 8;
      ctxh[((size_t)(b * SEQ + q)) * DMODEL + h * DK + n] = (_Float16)cc[nt][r];
    }
  }
}

// ---------------------------------------------------------------------------
// Kernel 5: out = ctx @ Wo^T + bo  (f32 output).  Same blocking as kernel 1.
// ---------------------------------------------------------------------------
__global__ void mha_outproj_kernel(const _Float16* __restrict__ ctxh,
                                   const _Float16* __restrict__ Woh,
                                   const float* __restrict__ bo,
                                   float* __restrict__ out) {
  const int lane = threadIdx.x & 31;
  const int wave = threadIdx.x >> 5;
  const int m0 = blockIdx.y * 128 + (wave >> 1) * 32;
  const int n0 = blockIdx.x * 64 + (wave & 1) * 32;
  const int mr = lane & 15;
  const int half = lane >> 4;

  const _Float16* a0 = ctxh + (size_t)(m0 + mr) * DMODEL + half * 8;
  const _Float16* a1 = a0 + (size_t)16 * DMODEL;
  const _Float16* b0 = Woh + (size_t)(n0 + mr) * DMODEL + half * 8;
  const _Float16* b1 = b0 + (size_t)16 * DMODEL;

  v8f c00 = {}, c01 = {}, c10 = {}, c11 = {};
#pragma unroll 2
  for (int k = 0; k < DMODEL; k += 32) {
    const v16h A0 = frag_f16(a0 + k), A1 = frag_f16(a1 + k);
    const v16h B0 = frag_f16(b0 + k), B1 = frag_f16(b1 + k);
    c00 = wmma16(A0, B0, c00);
    c01 = wmma16(A0, B1, c01);
    c10 = wmma16(A1, B0, c10);
    c11 = wmma16(A1, B1, c11);
  }

  v8f cc[4] = {c00, c01, c10, c11};
#pragma unroll
  for (int ti = 0; ti < 4; ++ti) {
    const int mb = m0 + ((ti >> 1) ? 16 : 0);
    const int n = n0 + ((ti & 1) ? 16 : 0) + mr;
    const float bv = bo[n];
#pragma unroll
    for (int r = 0; r < 8; ++r)
      out[(size_t)(mb + r + half * 8) * DMODEL + n] = cc[ti][r] + bv;
  }
}

// ---------------------------------------------------------------------------
extern "C" void kernel_launch(void* const* d_in, const int* in_sizes, int n_in,
                              void* d_out, int out_size, void* d_ws, size_t ws_size,
                              hipStream_t stream) {
  const float* Q  = (const float*)d_in[0];
  const float* K  = (const float*)d_in[1];
  const float* V  = (const float*)d_in[2];
  // d_in[3] = causal mask (int) -- implemented analytically, not read.
  const float* Wq = (const float*)d_in[4];
  const float* bq = (const float*)d_in[5];
  const float* Wk = (const float*)d_in[6];
  const float* bk = (const float*)d_in[7];
  const float* Wv = (const float*)d_in[8];
  const float* bv = (const float*)d_in[9];
  const float* Wo = (const float*)d_in[10];
  const float* bo = (const float*)d_in[11];

  float* out  = (float*)d_out;
  float* attn = out + OUT0;                       // 1 GB attn region

  // Workspace carve-up (f16 elements), total 104 MB:
  //   Qf|Kf|Vf : packed inputs   (3 x 16 MB)
  //   Wqh..Woh : packed weights  (4 x 2 MB)
  //   qh|kh|vT : projections     (3 x 16 MB)
  //   ctxh     : aliases Qf (dead after its projection; stream-ordered)
  const size_t XN = (size_t)MROWS * DMODEL;       // 8M elements
  const size_t WN = (size_t)DMODEL * DMODEL;      // 1M elements
  _Float16* Qf  = (_Float16*)d_ws;
  _Float16* Kf  = Qf  + XN;
  _Float16* Vf  = Kf  + XN;
  _Float16* Wqh = Vf  + XN;
  _Float16* Wkh = Wqh + WN;
  _Float16* Wvh = Wkh + WN;
  _Float16* Woh = Wvh + WN;
  _Float16* qh  = Woh + WN;
  _Float16* kh  = qh  + XN;
  _Float16* vT  = kh  + XN;
  _Float16* ctxh = Qf;

  const dim3 blk(256);

  // Pack to f16.
  mha_cvt_kernel<<<dim3((int)(XN / 4 / 256)), blk, 0, stream>>>(Q, Qf, (int)(XN / 4));
  mha_cvt_kernel<<<dim3((int)(XN / 4 / 256)), blk, 0, stream>>>(K, Kf, (int)(XN / 4));
  mha_cvt_kernel<<<dim3((int)(XN / 4 / 256)), blk, 0, stream>>>(V, Vf, (int)(XN / 4));
  mha_cvt_kernel<<<dim3((int)(WN / 4 / 256)), blk, 0, stream>>>(Wq, Wqh, (int)(WN / 4));
  mha_cvt_kernel<<<dim3((int)(WN / 4 / 256)), blk, 0, stream>>>(Wk, Wkh, (int)(WN / 4));
  mha_cvt_kernel<<<dim3((int)(WN / 4 / 256)), blk, 0, stream>>>(Wv, Wvh, (int)(WN / 4));
  mha_cvt_kernel<<<dim3((int)(WN / 4 / 256)), blk, 0, stream>>>(Wo, Woh, (int)(WN / 4));

  // Projections (q/k head-split, v transposed).
  const dim3 gProj(DMODEL / 64, MROWS / 128, 1);
  mha_proj_kernel<<<gProj, blk, 0, stream>>>(Qf, Wqh, bq, qh, 0);
  mha_proj_kernel<<<gProj, blk, 0, stream>>>(Kf, Wkh, bk, kh, 0);
  mha_proj_kernel<<<gProj, blk, 0, stream>>>(Vf, Wvh, bv, vT, 1);

  // Attention.
  mha_scores_kernel<<<dim3(SEQ / 64, SEQ / 128, BATCH * NHEADS), blk, 0, stream>>>(
      qh, kh, attn);
  mha_softmax_kernel<<<dim3(BATCH * NHEADS * SEQ), blk, 0, stream>>>(attn);
  mha_ctx_kernel<<<dim3(1, SEQ / 128, BATCH * NHEADS), blk, 0, stream>>>(
      attn, vT, ctxh);

  // Output projection.
  mha_outproj_kernel<<<gProj, blk, 0, stream>>>(ctxh, Woh, bo, out);
}
